// GraphSageLinkPredNoEmb_56624848830739
// MI455X (gfx1250) — compile-verified, matrix operand order checked by reading, not run on hardware
//
#include <hip/hip_runtime.h>
#include <hip/hip_bf16.h>

typedef __attribute__((ext_vector_type(16))) __bf16 v16bf;
typedef __attribute__((ext_vector_type(8)))  float  v8f;

#define H128 128

// ---------------------------------------------------------------------------
// Fragment loader: 16 fp32 values at offsets {0..7, 16..23} from p, scaled,
// converted to bf16 in the VGPR order required by v_wmma_f32_16x16x32_bf16.
// (p already includes row*K + kchunk*32 + (lane>>4)*8.)
// ---------------------------------------------------------------------------
__device__ __forceinline__ v16bf load_frag_bf16(const float* __restrict__ p, float s) {
    const float4 f0 = *(const float4*)(p + 0);
    const float4 f1 = *(const float4*)(p + 4);
    const float4 f2 = *(const float4*)(p + 16);
    const float4 f3 = *(const float4*)(p + 20);
    v16bf r;
    r[0]  = (__bf16)(f0.x * s); r[1]  = (__bf16)(f0.y * s);
    r[2]  = (__bf16)(f0.z * s); r[3]  = (__bf16)(f0.w * s);
    r[4]  = (__bf16)(f1.x * s); r[5]  = (__bf16)(f1.y * s);
    r[6]  = (__bf16)(f1.z * s); r[7]  = (__bf16)(f1.w * s);
    r[8]  = (__bf16)(f2.x * s); r[9]  = (__bf16)(f2.y * s);
    r[10] = (__bf16)(f2.z * s); r[11] = (__bf16)(f2.w * s);
    r[12] = (__bf16)(f3.x * s); r[13] = (__bf16)(f3.y * s);
    r[14] = (__bf16)(f3.z * s); r[15] = (__bf16)(f3.w * s);
    return r;
}

// ---------------------------------------------------------------------------
// C[M,128] = relu?( scale_rows(A1,1/max(deg1,1)) @ W1^T + A2 @ W2^T + bias )
// W row-major [128, K]; A2/deg1/bias may be null.
// block=(32,8): 8 waves = 4 row-tiles x 2 col-halves; each wave computes a
// 16x64 output tile (4 accumulators) reusing one A fragment per k-chunk.
// grid.x = ceil(M/64).
// ---------------------------------------------------------------------------
__global__ void gemm_dual_wmma(const float* __restrict__ A1, const float* __restrict__ W1,
                               const float* __restrict__ deg1,
                               const float* __restrict__ A2, const float* __restrict__ W2,
                               const float* __restrict__ bias,
                               float* __restrict__ C,
                               int M, int K1, int K2, int relu) {
    const int lane = threadIdx.x;          // 0..31
    const int wy   = threadIdx.y & 3;      // row tile within block (16 rows)
    const int wx   = threadIdx.y >> 2;     // col half (64 cols)
    const int rlo  = lane & 15;
    const int hi   = lane >> 4;
    const int row_g = blockIdx.x * 64 + wy * 16 + rlo;
    const int row_c = (row_g < M) ? row_g : (M - 1);
    const int col0  = wx * 64 + rlo;       // col of strip j is col0 + 16*j

    float s1 = 1.0f;
    if (deg1) s1 = 1.0f / fmaxf(deg1[row_c], 1.0f);

    v8f acc[4];
#pragma unroll
    for (int j = 0; j < 4; ++j) {
        const float bv = bias ? bias[col0 + 16 * j] : 0.0f;
#pragma unroll
        for (int i = 0; i < 8; ++i) acc[j][i] = bv;
    }

    {
        const float* arow = A1 + (long)row_c * K1 + hi * 8;
        const int nk = K1 >> 5;
        for (int kc = 0; kc < nk; ++kc) {
            const v16bf af = load_frag_bf16(arow + kc * 32, s1);
            v16bf bfr[4];
#pragma unroll
            for (int j = 0; j < 4; ++j)
                bfr[j] = load_frag_bf16(W1 + (long)(col0 + 16 * j) * K1 + kc * 32 + hi * 8, 1.0f);
#pragma unroll
            for (int j = 0; j < 4; ++j)
                acc[j] = __builtin_amdgcn_wmma_f32_16x16x32_bf16(false, af, false, bfr[j],
                                                                (short)0, acc[j], false, false);
        }
    }
    if (A2) {
        const float* arow = A2 + (long)row_c * K2 + hi * 8;
        const int nk = K2 >> 5;
        for (int kc = 0; kc < nk; ++kc) {
            const v16bf af = load_frag_bf16(arow + kc * 32, 1.0f);
            v16bf bfr[4];
#pragma unroll
            for (int j = 0; j < 4; ++j)
                bfr[j] = load_frag_bf16(W2 + (long)(col0 + 16 * j) * K2 + kc * 32 + hi * 8, 1.0f);
#pragma unroll
            for (int j = 0; j < 4; ++j)
                acc[j] = __builtin_amdgcn_wmma_f32_16x16x32_bf16(false, af, false, bfr[j],
                                                                (short)0, acc[j], false, false);
        }
    }

    // C/D layout: VGPR r -> row (r + 8*hi), col = lane%16 within 16x16 tile.
#pragma unroll
    for (int j = 0; j < 4; ++j) {
#pragma unroll
        for (int r = 0; r < 8; ++r) {
            float v = acc[j][r];
            if (relu) v = fmaxf(v, 0.0f);
            const int rowO = blockIdx.x * 64 + wy * 16 + hi * 8 + r;
            if (rowO < M) C[(long)rowO * H128 + col0 + 16 * j] = v;
        }
    }
}

// ---------------------------------------------------------------------------
// zero a float buffer (n divisible by 4)
// ---------------------------------------------------------------------------
__global__ void zero_f32(float* __restrict__ p, long n4) {
    long i = (long)blockIdx.x * blockDim.x + threadIdx.x;
    float4 z = make_float4(0.f, 0.f, 0.f, 0.f);
    for (; i < n4; i += (long)gridDim.x * blockDim.x) ((float4*)p)[i] = z;
}

// ---------------------------------------------------------------------------
// degree counters: deg_p[edge_dst[e]] += 1 ; deg_u[edge_src[e]] += 1
// ---------------------------------------------------------------------------
__global__ void degree_kernel(const int* __restrict__ esrc, const int* __restrict__ edst,
                              float* __restrict__ deg_u, float* __restrict__ deg_p, int E) {
    int e = blockIdx.x * blockDim.x + threadIdx.x;
    if (e < E) {
        atomicAdd(&deg_p[edst[e]], 1.0f);
        atomicAdd(&deg_u[esrc[e]], 1.0f);
    }
}

// ---------------------------------------------------------------------------
// Agg[dst[e], :] += X[src[e], :]   (wave per edge, float4 per lane)
// ---------------------------------------------------------------------------
__global__ void scatter_add_kernel(const float* __restrict__ X,
                                   const int* __restrict__ src, const int* __restrict__ dst,
                                   float* __restrict__ Agg, int E) {
    const int wave = (blockIdx.x * blockDim.x + threadIdx.x) >> 5;
    const int lane = threadIdx.x & 31;
    if (wave >= E) return;
    const int s = src[wave], d = dst[wave];
    const float4 v = ((const float4*)(X + (long)s * H128))[lane];
    float* a = Agg + (long)d * H128 + lane * 4;
    atomicAdd(a + 0, v.x);
    atomicAdd(a + 1, v.y);
    atomicAdd(a + 2, v.z);
    atomicAdd(a + 3, v.w);
}

// ---------------------------------------------------------------------------
// out[i] = dot(xu[label_src[i]], xp[label_dst[i]])  (wave per pair, wave32)
// ---------------------------------------------------------------------------
__global__ void link_dot_kernel(const float* __restrict__ xu, const float* __restrict__ xp,
                                const int* __restrict__ ls, const int* __restrict__ ld,
                                float* __restrict__ out, int L) {
    const int wave = (blockIdx.x * blockDim.x + threadIdx.x) >> 5;
    const int lane = threadIdx.x & 31;
    if (wave >= L) return;
    const float4 a = ((const float4*)(xu + (long)ls[wave] * H128))[lane];
    const float4 b = ((const float4*)(xp + (long)ld[wave] * H128))[lane];
    float p = a.x * b.x + a.y * b.y + a.z * b.z + a.w * b.w;
#pragma unroll
    for (int off = 16; off > 0; off >>= 1) p += __shfl_xor(p, off, 32);
    if (lane == 0) out[wave] = p;
}

// ---------------------------------------------------------------------------
extern "C" void kernel_launch(void* const* d_in, const int* in_sizes, int n_in,
                              void* d_out, int out_size, void* d_ws, size_t ws_size,
                              hipStream_t stream) {
    const float* x_user  = (const float*)d_in[0];
    const float* x_prod  = (const float*)d_in[1];
    const int*   esrc    = (const int*)  d_in[2];
    const int*   edst    = (const int*)  d_in[3];
    const int*   lsrc    = (const int*)  d_in[4];
    const int*   ldst    = (const int*)  d_in[5];
    const float* user_W  = (const float*)d_in[6];
    const float* user_b  = (const float*)d_in[7];
    const float* prod_W  = (const float*)d_in[8];
    const float* prod_b  = (const float*)d_in[9];
    const float* Wl_buy  = (const float*)d_in[10];
    const float* bl_buy  = (const float*)d_in[11];
    const float* Wr_buy  = (const float*)d_in[12];
    const float* Wl_rev  = (const float*)d_in[13];
    const float* bl_rev  = (const float*)d_in[14];
    const float* Wr_rev  = (const float*)d_in[15];
    float* out = (float*)d_out;

    const int Hh = in_sizes[7];                 // 128
    const int FU = in_sizes[6] / Hh;            // 64
    const int FP = in_sizes[8] / Hh;            // 64
    const int NU = in_sizes[0] / FU;            // 100000
    const int NP = in_sizes[1] / FP;            // 50000
    const int E  = in_sizes[2];                 // 600000
    const int L  = in_sizes[4];                 // 200000
    const int NL = in_sizes[11] / Hh;           // 3

    // workspace layout (floats)
    float* ws = (float*)d_ws;
    float* xuA   = ws;                 ws += (long)NU * H128;
    float* xuB   = ws;                 ws += (long)NU * H128;
    float* xpA   = ws;                 ws += (long)NP * H128;
    float* xpB   = ws;                 ws += (long)NP * H128;
    float* agg_u = ws;                 ws += (long)NU * H128;
    float* agg_p = ws;                 ws += (long)NP * H128;
    float* deg_u = ws;                 ws += NU;
    float* deg_p = ws;                 ws += NP;
    (void)ws_size; (void)n_in; (void)out_size;

    const dim3 gblk(32, 8);            // 8 waves: 4 row-tiles x 2 col-halves

    // degrees
    zero_f32<<<512, 256, 0, stream>>>(deg_u, (long)NU / 4);
    zero_f32<<<512, 256, 0, stream>>>(deg_p, (long)NP / 4);
    degree_kernel<<<(E + 255) / 256, 256, 0, stream>>>(esrc, edst, deg_u, deg_p, E);

    // input projections: xu = x_user @ user_W^T + user_b ; xp likewise
    gemm_dual_wmma<<<dim3((NU + 63) / 64), gblk, 0, stream>>>(
        x_user, user_W, nullptr, nullptr, nullptr, user_b, xuA, NU, FU, 0, 0);
    gemm_dual_wmma<<<dim3((NP + 63) / 64), gblk, 0, stream>>>(
        x_prod, prod_W, nullptr, nullptr, nullptr, prod_b, xpA, NP, FP, 0, 0);

    float* xu_cur = xuA; float* xu_nxt = xuB;
    float* xp_cur = xpA; float* xp_nxt = xpB;

    const int nEdgeBlk = (E + 7) / 8;  // wave-per-edge, 8 waves/block
    for (int l = 0; l < NL; ++l) {
        zero_f32<<<2048, 256, 0, stream>>>(agg_p, (long)NP * H128 / 4);
        zero_f32<<<2048, 256, 0, stream>>>(agg_u, (long)NU * H128 / 4);
        // buy: user -> prod ; rev_buy: prod -> user
        scatter_add_kernel<<<nEdgeBlk, 256, 0, stream>>>(xu_cur, esrc, edst, agg_p, E);
        scatter_add_kernel<<<nEdgeBlk, 256, 0, stream>>>(xp_cur, edst, esrc, agg_u, E);

        const int relu = (l < NL - 1) ? 1 : 0;
        gemm_dual_wmma<<<dim3((NP + 63) / 64), gblk, 0, stream>>>(
            agg_p, Wl_buy + (long)l * Hh * Hh, deg_p,
            xp_cur, Wr_buy + (long)l * Hh * Hh, bl_buy + (long)l * Hh,
            xp_nxt, NP, Hh, Hh, relu);
        gemm_dual_wmma<<<dim3((NU + 63) / 64), gblk, 0, stream>>>(
            agg_u, Wl_rev + (long)l * Hh * Hh, deg_u,
            xu_cur, Wr_rev + (long)l * Hh * Hh, bl_rev + (long)l * Hh,
            xu_nxt, NU, Hh, Hh, relu);

        float* t;
        t = xu_cur; xu_cur = xu_nxt; xu_nxt = t;
        t = xp_cur; xp_cur = xp_nxt; xp_nxt = t;
    }

    link_dot_kernel<<<(L + 7) / 8, 256, 0, stream>>>(xu_cur, xp_cur, lsrc, ldst, out, L);
}
